// Net_35837207118569
// MI455X (gfx1250) — matrix operand order repaired and checked by reading, not compiled, not run on hardware
//
#include <hip/hip_runtime.h>
#include <hip/hip_bf16.h>
#include <math.h>

// ---------------- problem constants ----------------
#define NN   10000
#define EE   320000
#define KC   100
#define IND  128

typedef __attribute__((ext_vector_type(16))) __bf16       v16bf;
typedef __attribute__((ext_vector_type(8)))  float        v8f;
typedef __attribute__((ext_vector_type(4)))  unsigned int u32x4;

union bfrag { u32x4 q[2]; v16bf v; };   // 32 bytes = 16 bf16 = one WMMA fragment per lane

// ---------------- WMMA GEMM ----------------
// C[M x Npad] = act(A[M x K] @ B[K x Npad] + bias)
// A: row-major bf16 (M multiple of 16, K multiple of 32)
// Bsw: pre-swizzled bf16 fragments, layout [ktile][ntile16][lane][16]
// One 16x64 output tile per wave32; explicit double-buffered K-loop:
// loads for tile t+1 are issued before the 4 WMMAs of tile t.
template<int ACT, int OUTBF>
__global__ __launch_bounds__(256)
void wmma_gemm(const __bf16* __restrict__ A, const __bf16* __restrict__ Bsw,
               const float* __restrict__ bias, void* __restrict__ Cv,
               int M, int Npad, int K)
{
    const int wid  = (blockIdx.x << 3) + (threadIdx.x >> 5);  // 8 waves / block
    const int lane = threadIdx.x & 31;
    const int tilesN4 = Npad >> 6;                            // 64-wide tiles
    if (wid >= (M >> 4) * tilesN4) return;                    // wave-uniform exit

    const int tm = wid / tilesN4, tn = wid - tm * tilesN4;
    const int mlane = lane & 15;
    const int kh    = lane >> 4;                              // lane half
    const int tilesN16 = Npad >> 4;
    const int ktiles   = K >> 5;

    const __bf16* Arow  = A + (size_t)((tm << 4) + mlane) * K + (kh << 4);
    const __bf16* Bbase = Bsw + ((size_t)(tn << 2) * 32 + lane) * 16;
    const size_t  bstep = (size_t)tilesN16 * 512;             // elems per ktile

    auto loadA = [&](int t, bfrag& a) {
        const __bf16* ap = Arow + (t << 5);
        a.q[0] = *reinterpret_cast<const u32x4*>(ap);
        a.q[1] = *reinterpret_cast<const u32x4*>(ap + 8);
    };
    auto loadB = [&](int t, int c, bfrag& b) {
        const __bf16* bp = Bbase + (size_t)t * bstep + c * 512;
        b.q[0] = *reinterpret_cast<const u32x4*>(bp);
        b.q[1] = *reinterpret_cast<const u32x4*>(bp + 8);
    };

    v8f acc[4] = {};
    bfrag aCur, bCur[4];
    loadA(0, aCur);
    #pragma unroll
    for (int c = 0; c < 4; ++c) loadB(0, c, bCur[c]);

    for (int t = 0; t < ktiles; ++t) {
        const int tn1 = (t + 1 < ktiles) ? t + 1 : t;         // clamp: last iter reloads self
        __builtin_prefetch(Arow + (tn1 << 5) + 32, 0, 3);
        bfrag aNxt, bNxt[4];
        loadA(tn1, aNxt);
        #pragma unroll
        for (int c = 0; c < 4; ++c) loadB(tn1, c, bNxt[c]);
        #pragma unroll
        for (int c = 0; c < 4; ++c)
            acc[c] = __builtin_amdgcn_wmma_f32_16x16x32_bf16(false, aCur.v, false, bCur[c].v,
                                                             (short)0, acc[c], false, false);
        aCur = aNxt;
        #pragma unroll
        for (int c = 0; c < 4; ++c) bCur[c] = bNxt[c];
    }

    // C/D layout: VGPR r -> M = tm*16 + kh*8 + r, N = tn*64 + c*16 + mlane
    const int n0 = (tn << 6) + mlane;
    #pragma unroll
    for (int c = 0; c < 4; ++c) {
        const int n = n0 + (c << 4);
        const float bv = bias ? bias[n] : 0.f;
        #pragma unroll
        for (int r = 0; r < 8; ++r) {
            const int m = (tm << 4) + (kh << 3) + r;
            float v = acc[c][r] + bv;
            if (ACT) v = fmaxf(v, 0.f);
            if (OUTBF) ((__bf16*)Cv)[(size_t)m * Npad + n] = (__bf16)v;
            else       ((float*) Cv)[(size_t)m * Npad + n] = v;
        }
    }
}

// ---------------- staging kernels ----------------
// swizzle fp32 weight B[K x N] -> bf16 fragment layout [K/32][Npad/16][lane 0..31][j 0..15]
// element (t,u,lane,j) = B[t*32 + (lane>>4)*16 + j][u*16 + (lane&15)], 0 for padded cols
__global__ void k_swizzle_b(const float* __restrict__ B, __bf16* __restrict__ Bsw,
                            int K, int N, int Npad) {
    int idx = blockIdx.x * blockDim.x + threadIdx.x;
    int tilesN16 = Npad >> 4;
    int total = (K >> 5) * tilesN16 * 512;
    if (idx >= total) return;
    int j    = idx & 15;
    int lane = (idx >> 4) & 31;
    int u    = (idx >> 9) % tilesN16;
    int t    = (idx >> 9) / tilesN16;
    int k = (t << 5) + ((lane >> 4) << 4) + j;
    int n = (u << 4) + (lane & 15);
    float v = (n < N) ? B[(size_t)k * N + n] : 0.f;
    Bsw[idx] = (__bf16)v;
}

// fp32 [rows_valid x F] (ld=ldin) -> bf16 row-major [rows_total x F], pad rows zeroed
__global__ void k_cvt_bf16(const float* __restrict__ in, __bf16* __restrict__ out,
                           int rows_valid, int rows_total, int F, int ldin) {
    int idx = blockIdx.x * blockDim.x + threadIdx.x;
    if (idx >= rows_total * F) return;
    int i = idx / F, f = idx - i * F;
    float v = (i < rows_valid) ? in[(size_t)i * ldin + f] : 0.f;
    out[idx] = (__bf16)v;
}

__global__ void k_padvec(const float* __restrict__ in, float* __restrict__ out,
                         int nvalid, int ntot) {
    int i = blockIdx.x * blockDim.x + threadIdx.x;
    if (i < ntot) out[i] = (i < nvalid) ? in[i] : 0.f;
}

// ---------------- graph kernels ----------------
__global__ void k_deg(const int* __restrict__ col, const float* __restrict__ ew,
                      float* __restrict__ deg, int E) {
    int e = blockIdx.x * blockDim.x + threadIdx.x;
    if (e < E) atomicAdd(&deg[col[e]], ew[e]);
}

__global__ void k_dinv(const float* __restrict__ deg, float* __restrict__ dinv, int n) {
    int i = blockIdx.x * blockDim.x + threadIdx.x;
    if (i < n) dinv[i] = rsqrtf(deg[i] + 1.0f);
}

// h[i,f] = dinv[i]^2 * t[i,f] + bias[f]
__global__ void k_selfloop_bias(const float* __restrict__ t, const float* __restrict__ dinv,
                                const float* __restrict__ bias, float* __restrict__ h,
                                int n, int F) {
    int idx = blockIdx.x * blockDim.x + threadIdx.x;
    if (idx >= n * F) return;
    int i = idx / F, f = idx - i * F;
    float d = dinv[i];
    h[idx] = d * d * t[idx] + bias[f];
}

// h[col[e],f] += dinv[row]*ew*dinv[col] * t[row[e],f]  (4 features / thread)
__global__ void k_edge_scatter(const int* __restrict__ row, const int* __restrict__ col,
                               const float* __restrict__ ew, const float* __restrict__ dinv,
                               const float* __restrict__ t, float* __restrict__ h,
                               int E, int F) {
    int idx = blockIdx.x * blockDim.x + threadIdx.x;
    int per = F >> 2;
    int e = idx / per;
    if (e >= E) return;
    int f = (idx - e * per) << 2;
    int r = row[e], c = col[e];
    float nrm = dinv[r] * ew[e] * dinv[c];
    const float* tp = t + (size_t)r * F + f;
    float* hp = h + (size_t)c * F + f;
    atomicAdd(hp + 0, nrm * tp[0]);
    atomicAdd(hp + 1, nrm * tp[1]);
    atomicAdd(hp + 2, nrm * tp[2]);
    atomicAdd(hp + 3, nrm * tp[3]);
}

__global__ void k_relu(float* __restrict__ x, int n) {
    int i = blockIdx.x * blockDim.x + threadIdx.x;
    if (i < n) x[i] = fmaxf(x[i], 0.f);
}

// per-column (axis=0) softmax stats over logits [rows x kc] (ld)
__global__ void k_colstats(const float* __restrict__ logits, float* __restrict__ colmax,
                           float* __restrict__ collog, int kc, int rows, int ld) {
    __shared__ float red[256];
    int k = blockIdx.x;
    float m = -3.4e38f;
    for (int i = threadIdx.x; i < rows; i += blockDim.x)
        m = fmaxf(m, logits[(size_t)i * ld + k]);
    red[threadIdx.x] = m; __syncthreads();
    for (int s = 128; s > 0; s >>= 1) {
        if (threadIdx.x < s) red[threadIdx.x] = fmaxf(red[threadIdx.x], red[threadIdx.x + s]);
        __syncthreads();
    }
    float cm = red[0]; __syncthreads();
    float sum = 0.f;
    for (int i = threadIdx.x; i < rows; i += blockDim.x)
        sum += __expf(logits[(size_t)i * ld + k] - cm);
    red[threadIdx.x] = sum; __syncthreads();
    for (int s = 128; s > 0; s >>= 1) {
        if (threadIdx.x < s) red[threadIdx.x] += red[threadIdx.x + s];
        __syncthreads();
    }
    if (threadIdx.x == 0) { colmax[k] = cm; collog[k] = __logf(red[0]); }
}

// per-row argmax of column-softmaxed logits -> one-hot map (ties kept),
// primary cluster id, per-cluster counts (= diag(map^T map)), loss2 accumulation
__global__ void k_rowmap(const float* __restrict__ logits, const float* __restrict__ colmax,
                         const float* __restrict__ collog, float* __restrict__ map_out,
                         int* __restrict__ cluster, float* __restrict__ counts,
                         float* __restrict__ l2acc, int rows, int kc, int ld) {
    int i = blockIdx.x * blockDim.x + threadIdx.x;
    if (i >= rows) return;
    float best = -3.4e38f; int bi = 0;
    for (int k = 0; k < kc; ++k) {
        float z = logits[(size_t)i * ld + k] - colmax[k] - collog[k];
        if (z > best) { best = z; bi = k; }
    }
    float ones = 0.f;
    for (int k = 0; k < kc; ++k) {
        float z = logits[(size_t)i * ld + k] - colmax[k] - collog[k];
        float m = (z == best) ? 1.f : 0.f;
        map_out[(size_t)i * kc + k] = m;
        if (m != 0.f) { ones += 1.f; atomicAdd(&counts[k], 1.f); }
    }
    cluster[i] = bi;
    atomicAdd(l2acc, ones * (-__logf(1.0f + 1e-6f)));
}

// region[k,:] += h2[i,:] for every (i,k) with map==1 (ties honoured)
__global__ void k_region_scatter(const float* __restrict__ map_out, const float* __restrict__ h2,
                                 float* __restrict__ region, int rows, int kc, int F) {
    int idx = blockIdx.x * blockDim.x + threadIdx.x;
    if (idx >= rows * kc) return;
    if (map_out[idx] == 0.f) return;
    int i = idx / kc, k = idx - i * kc;
    const float* hp = h2 + (size_t)i * F;
    float* rp = region + (size_t)k * F;
    for (int f = 0; f < F; ++f) atomicAdd(rp + f, hp[f]);
}

__global__ void k_outadj(const int* __restrict__ row, const int* __restrict__ col,
                         const float* __restrict__ ew, const int* __restrict__ cluster,
                         float* __restrict__ out_adj, int E, int kc) {
    int e = blockIdx.x * blockDim.x + threadIdx.x;
    if (e >= E) return;
    atomicAdd(&out_adj[(size_t)cluster[row[e]] * kc + cluster[col[e]]], ew[e]);
}

__global__ void k_finalize(const float* __restrict__ counts, const float* __restrict__ l2acc,
                           float* __restrict__ out_l1, float* __restrict__ out_l2,
                           int kc, float inv_sqrt_nk, float inv_n) {
    __shared__ float red[128];
    float s = 0.f;
    for (int k = threadIdx.x; k < kc; k += blockDim.x) s += sqrtf(counts[k] + 1e-15f);
    red[threadIdx.x] = s; __syncthreads();
    for (int st = 64; st > 0; st >>= 1) {
        if (threadIdx.x < st) red[threadIdx.x] += red[threadIdx.x + st];
        __syncthreads();
    }
    if (threadIdx.x == 0) { out_l1[0] = -red[0] * inv_sqrt_nk; out_l2[0] = l2acc[0] * inv_n; }
}

__global__ void k_adjfix(const float* __restrict__ out_adj, float* __restrict__ adj2, int kc) {
    int idx = blockIdx.x * blockDim.x + threadIdx.x;
    if (idx >= kc * kc) return;
    int i = idx / kc, j = idx - i * kc;
    adj2[idx] = (i == j) ? 1.f : out_adj[idx];
}

__global__ void k_colsum_rsqrt(const float* __restrict__ adj2, float* __restrict__ dr, int kc) {
    int k = blockIdx.x * blockDim.x + threadIdx.x;
    if (k >= kc) return;
    float s = 0.f;
    for (int j = 0; j < kc; ++j) s += adj2[(size_t)j * kc + k];
    dr[k] = rsqrtf(s);
}

// out[i,f] = act(dr[i] * sum_j adj2[j,i]*dr[j]*X[j,f] + bias[f])
// writes bf16 (next-GEMM operand) and optionally fp32 (final output)
template<int ACT>
__global__ void k_pooled_prop(const float* __restrict__ adj2, const float* __restrict__ dr,
                              const float* __restrict__ X, const float* __restrict__ bias,
                              __bf16* __restrict__ outbf, float* __restrict__ outf,
                              int kc, int F) {
    int idx = blockIdx.x * blockDim.x + threadIdx.x;
    if (idx >= kc * F) return;
    int i = idx / F, f = idx - i * F;
    float acc = 0.f;
    for (int j = 0; j < kc; ++j)
        acc += adj2[(size_t)j * kc + i] * dr[j] * X[(size_t)j * F + f];
    float v = dr[i] * acc + bias[f];
    if (ACT) v = fmaxf(v, 0.f);
    outbf[idx] = (__bf16)v;
    if (outf) outf[idx] = v;
}

__global__ void k_pred(const float* __restrict__ p2, const float* __restrict__ pw3,
                       const float* __restrict__ pb3, float* __restrict__ out, int kc, int F) {
    int i = blockIdx.x * blockDim.x + threadIdx.x;
    if (i >= kc) return;
    float acc = 0.f;
    for (int j = 0; j < F; ++j) acc += p2[(size_t)i * F + j] * pw3[j];
    out[i] = acc + pb3[0];
}

// ---------------- host orchestration ----------------
static inline void launch_gemm(int act, int outbf,
                               const __bf16* A, const __bf16* Bsw, const float* bias,
                               void* C, int M, int Npad, int K, hipStream_t s) {
    int waves = (M >> 4) * (Npad >> 6);
    int blocks = (waves + 7) / 8;
    if (act) {
        if (outbf) wmma_gemm<1,1><<<blocks, 256, 0, s>>>(A, Bsw, bias, C, M, Npad, K);
        else       wmma_gemm<1,0><<<blocks, 256, 0, s>>>(A, Bsw, bias, C, M, Npad, K);
    } else {
        wmma_gemm<0,0><<<blocks, 256, 0, s>>>(A, Bsw, bias, C, M, Npad, K);
    }
}

static inline void launch_swizzle(const float* B, __bf16* Bsw, int K, int N, int Npad,
                                  hipStream_t s) {
    int total = (K >> 5) * (Npad >> 4) * 512;
    k_swizzle_b<<<(total + 255) / 256, 256, 0, s>>>(B, Bsw, K, N, Npad);
}

extern "C" void kernel_launch(void* const* d_in, const int* in_sizes, int n_in,
                              void* d_out, int out_size, void* d_ws, size_t ws_size,
                              hipStream_t stream) {
    const float* x   = (const float*)d_in[0];
    const int*  eidx = (const int*)  d_in[1];
    const float* ew  = (const float*)d_in[2];
    const float* w1  = (const float*)d_in[3];
    const float* b1  = (const float*)d_in[4];
    const float* w2  = (const float*)d_in[5];
    const float* b2  = (const float*)d_in[6];
    const float* aw1 = (const float*)d_in[7];
    const float* ab1 = (const float*)d_in[8];
    const float* aw2 = (const float*)d_in[9];
    const float* ab2 = (const float*)d_in[10];
    const float* w3  = (const float*)d_in[11];
    const float* b3  = (const float*)d_in[12];
    const float* w4  = (const float*)d_in[13];
    const float* b4  = (const float*)d_in[14];
    const float* pw1 = (const float*)d_in[15];
    const float* pb1 = (const float*)d_in[16];
    const float* pw2 = (const float*)d_in[17];
    const float* pb2 = (const float*)d_in[18];
    const float* pw3 = (const float*)d_in[19];
    const float* pb3 = (const float*)d_in[20];
    const int* row = eidx;
    const int* col = eidx + EE;

    // ---- workspace carving (float units, 64-float = 256B aligned) ----
    float* ws = (float*)d_ws;
    size_t off = 0;
    auto alloc   = [&](size_t nfloats) { size_t o = off; off += (nfloats + 63) & ~(size_t)63; return o; };
    auto allocbf = [&](size_t nelems)  { return alloc((nelems + 1) / 2); };

    float* DEG    = ws + alloc(NN);
    float* DINV   = ws + alloc(NN);
    float* T      = ws + alloc((size_t)NN * 512);   // fp32 t1 / t2
    float* H1     = ws + alloc((size_t)NN * 256);   // fp32 h1, later logits (ld 128)
    float* H2     = ws + alloc((size_t)NN * 256);   // fp32 h2
    float* COLMAX = ws + alloc(128);
    float* COLLOG = ws + alloc(128);
    float* COUNTS = ws + alloc(128);
    float* L2ACC  = ws + alloc(64);
    float* REGION = ws + alloc(KC * 256);
    float* OUTADJ = ws + alloc(KC * KC);
    float* ADJ2   = ws + alloc(KC * KC);
    float* DR     = ws + alloc(128);
    float* HH     = ws + alloc(112 * 512);          // fp32 region@w3
    float* HH2    = ws + alloc(112 * 256);          // fp32 xr1@w4
    float* P2     = ws + alloc(112 * 512);          // fp32 p2
    float* AB2P   = ws + alloc(128);                // padded ab2 (128)
    int*   CLUST  = (int*)(ws + alloc(NN));

    __bf16* XBF  = (__bf16*)(ws + allocbf((size_t)NN * IND));
    __bf16* HBF  = (__bf16*)(ws + allocbf((size_t)NN * 256));
    __bf16* H2BF = (__bf16*)(ws + allocbf((size_t)NN * 256));
    __bf16* ABF  = (__bf16*)(ws + allocbf((size_t)NN * 512));  // bf16 'a'
    __bf16* SREG = (__bf16*)(ws + allocbf(112 * 256));
    __bf16* SX1  = (__bf16*)(ws + allocbf(112 * 512));
    __bf16* SXC  = (__bf16*)(ws + allocbf(112 * 256));
    __bf16* SP1  = (__bf16*)(ws + allocbf(112 * 512));
    __bf16* BW1  = (__bf16*)(ws + allocbf((IND/32) * (256/16) * 512));
    __bf16* BW2  = (__bf16*)(ws + allocbf((256/32) * (256/16) * 512));
    __bf16* BAW1 = (__bf16*)(ws + allocbf((256/32) * (512/16) * 512));
    __bf16* BAW2 = (__bf16*)(ws + allocbf((512/32) * (128/16) * 512));
    __bf16* BW3  = (__bf16*)(ws + allocbf((256/32) * (512/16) * 512));
    __bf16* BW4  = (__bf16*)(ws + allocbf((512/32) * (256/16) * 512));
    __bf16* BPW1 = (__bf16*)(ws + allocbf((256/32) * (512/16) * 512));
    __bf16* BPW2 = (__bf16*)(ws + allocbf((512/32) * (512/16) * 512));

    float* out      = (float*)d_out;
    float* out_map  = out;                 // [NN, KC]
    float* out_l1   = out + 1000000;
    float* out_l2   = out + 1000001;
    float* out_xr   = out + 1000002;       // [KC, 256]
    float* out_pred = out + 1025602;       // [KC, 1]

    // ---- zero accumulators ----
    hipMemsetAsync(DEG,    0, NN * sizeof(float), stream);
    hipMemsetAsync(COUNTS, 0, 192 * sizeof(float), stream);                  // counts + l2acc
    hipMemsetAsync(REGION, 0, (KC * 256 + KC * KC) * sizeof(float), stream); // region + out_adj

    // ---- weight swizzles + staging ----
    launch_swizzle(w1,  BW1,  IND, 256, 256, stream);
    launch_swizzle(w2,  BW2,  256, 256, 256, stream);
    launch_swizzle(aw1, BAW1, 256, 512, 512, stream);
    launch_swizzle(aw2, BAW2, 512, KC,  128, stream);
    launch_swizzle(w3,  BW3,  256, 512, 512, stream);
    launch_swizzle(w4,  BW4,  512, 256, 256, stream);
    launch_swizzle(pw1, BPW1, 256, 512, 512, stream);
    launch_swizzle(pw2, BPW2, 512, 512, 512, stream);
    k_padvec<<<1, 128, 0, stream>>>(ab2, AB2P, KC, 128);
    k_cvt_bf16<<<(NN * IND + 255) / 256, 256, 0, stream>>>(x, XBF, NN, NN, IND, IND);

    // ---- degrees ----
    k_deg <<<(EE + 255) / 256, 256, 0, stream>>>(col, ew, DEG, EE);
    k_dinv<<<(NN + 255) / 256, 256, 0, stream>>>(DEG, DINV, NN);

    // ---- layer 1: h1 = relu(gcn_sparse(x, w1, b1)) ----
    launch_gemm(0, 0, XBF, BW1, nullptr, T, NN, 256, IND, stream);
    k_selfloop_bias<<<(NN * 256 + 255) / 256, 256, 0, stream>>>(T, DINV, b1, H1, NN, 256);
    k_edge_scatter<<<(EE * 64 + 255) / 256, 256, 0, stream>>>(row, col, ew, DINV, T, H1, EE, 256);
    k_relu<<<(NN * 256 + 255) / 256, 256, 0, stream>>>(H1, NN * 256);
    k_cvt_bf16<<<(NN * 256 + 255) / 256, 256, 0, stream>>>(H1, HBF, NN, NN, 256, 256);

    // ---- layer 2: h2 = gcn_sparse(h1, w2, b2) ----
    launch_gemm(0, 0, HBF, BW2, nullptr, T, NN, 256, 256, stream);
    k_selfloop_bias<<<(NN * 256 + 255) / 256, 256, 0, stream>>>(T, DINV, b2, H2, NN, 256);
    k_edge_scatter<<<(EE * 64 + 255) / 256, 256, 0, stream>>>(row, col, ew, DINV, T, H2, EE, 256);
    k_cvt_bf16<<<(NN * 256 + 255) / 256, 256, 0, stream>>>(H2, H2BF, NN, NN, 256, 256);

    // ---- assignment: a = relu(h2@aw1+ab1) (bf16 direct); logits = a@aw2+ab2 (ld 128) ----
    launch_gemm(1, 1, H2BF, BAW1, ab1, ABF, NN, 512, 256, stream);
    launch_gemm(0, 0, ABF, BAW2, AB2P, H1, NN, 128, 512, stream);

    // ---- softmax(axis=0) stats + row argmax -> map / cluster / counts / loss2 ----
    k_colstats<<<KC, 256, 0, stream>>>(H1, COLMAX, COLLOG, KC, NN, 128);
    k_rowmap<<<(NN + 255) / 256, 256, 0, stream>>>(H1, COLMAX, COLLOG, out_map,
                                                   CLUST, COUNTS, L2ACC, NN, KC, 128);

    // ---- pooling ----
    k_region_scatter<<<(NN * KC + 255) / 256, 256, 0, stream>>>(out_map, H2, REGION, NN, KC, 256);
    k_outadj<<<(EE + 255) / 256, 256, 0, stream>>>(row, col, ew, CLUST, OUTADJ, EE, KC);
    k_finalize<<<1, 128, 0, stream>>>(COUNTS, L2ACC, out_l1, out_l2, KC,
                                      1.0f / sqrtf((float)NN * (float)KC), 1.0f / (float)NN);
    k_adjfix<<<(KC * KC + 255) / 256, 256, 0, stream>>>(OUTADJ, ADJ2, KC);
    k_colsum_rsqrt<<<1, 128, 0, stream>>>(ADJ2, DR, KC);
    k_cvt_bf16<<<(112 * 256 + 255) / 256, 256, 0, stream>>>(REGION, SREG, KC, 112, 256, 256);

    // ---- region GCN 1: xr1 = relu(gcn_dense(region, w3, b3)) ----
    launch_gemm(0, 0, SREG, BW3, nullptr, HH, 112, 512, 256, stream);
    k_pooled_prop<1><<<(KC * 512 + 255) / 256, 256, 0, stream>>>(ADJ2, DR, HH, b3,
                                                                 SX1, nullptr, KC, 512);
    // ---- region GCN 2: xr = gcn_dense(xr1, w4, b4) (fp32 -> d_out, bf16 -> MLP) ----
    launch_gemm(0, 0, SX1, BW4, nullptr, HH2, 112, 256, 512, stream);
    k_pooled_prop<0><<<(KC * 256 + 255) / 256, 256, 0, stream>>>(ADJ2, DR, HH2, b4,
                                                                 SXC, out_xr, KC, 256);

    // ---- MLP readout ----
    launch_gemm(1, 1, SXC, BPW1, pb1, SP1, 112, 512, 256, stream);
    launch_gemm(1, 0, SP1, BPW2, pb2, P2, 112, 512, 512, stream);
    k_pred<<<1, 128, 0, stream>>>(P2, pw3, pb3, out_pred, KC, 512);

    (void)in_sizes; (void)n_in; (void)out_size; (void)ws_size;
}
